// ClassicInteractionBlock_61418032333204
// MI455X (gfx1250) — matrix operand order repaired and checked by reading, not compiled, not run on hardware
//
#include <hip/hip_runtime.h>

typedef __attribute__((ext_vector_type(16))) __bf16 v16bf;
typedef __attribute__((ext_vector_type(8)))  __bf16 v8bf;
typedef __attribute__((ext_vector_type(8)))  float  v8f;

// Branch-free fast tanh: 1 - 2/(exp(2x)+1).
// exp overflow -> +1, underflow -> -1; no divergent libm slow path.
__device__ __forceinline__ float fast_tanh(float x) {
    const float e = __expf(2.0f * x);                       // v_exp_f32 path
    return 1.0f - 2.0f * __builtin_amdgcn_rcpf(e + 1.0f);   // v_rcp_f32
}

// ---------------------------------------------------------------------------
// Weight conversion: f32 row-major W[K,128] -> bf16 transposed Wt[128,K]
// ---------------------------------------------------------------------------
__global__ void convert_weights(const float* __restrict__ Wi,  const float* __restrict__ W1,
                                const float* __restrict__ W2,  const float* __restrict__ Wo1,
                                const float* __restrict__ Wo2,
                                __bf16* __restrict__ WiT, __bf16* __restrict__ W1T,
                                __bf16* __restrict__ W2T, __bf16* __restrict__ Wo1T,
                                __bf16* __restrict__ Wo2T)
{
    int i = blockIdx.x * 256 + threadIdx.x;
    const float* src; __bf16* dst; int Kd;
    if      (i < 16384) {              src = Wi;  dst = WiT;  Kd = 128; }
    else if (i < 24576) { i -= 16384;  src = W1;  dst = W1T;  Kd = 64;  }
    else if (i < 40960) { i -= 24576;  src = W2;  dst = W2T;  Kd = 128; }
    else if (i < 57344) { i -= 40960;  src = Wo1; dst = Wo1T; Kd = 128; }
    else if (i < 73728) { i -= 57344;  src = Wo2; dst = Wo2T; Kd = 128; }
    else return;
    int n = i / Kd, k = i - n * Kd;
    dst[i] = (__bf16)src[k * 128 + n];   // N is always 128
}

// ---------------------------------------------------------------------------
// Generic bf16 WMMA GEMM: D[M,N] = act(A[M,K] @ B[K,N] + bias)
//  - Bt is B transposed, bf16, [N,K] row-major; each wave owns one 16x16 tile
// ---------------------------------------------------------------------------
template<bool A_BF16, bool OUT_BF16, bool DO_TANH, bool HAS_BIAS>
__global__ void __launch_bounds__(256)
gemm_wmma(const void* __restrict__ Av, const __bf16* __restrict__ Bt,
          const float* __restrict__ bias, void* __restrict__ Dv,
          int M, int N, int K)
{
    const int lane   = threadIdx.x & 31;
    const int lane15 = lane & 15;
    const int kh     = lane >> 4;
    const int tilesN = N >> 4;
    const int totalTiles = (M >> 4) * tilesN;
    const int tile = blockIdx.x * (blockDim.x >> 5) + (threadIdx.x >> 5);
    if (tile >= totalTiles) return;
    const int tM = tile / tilesN;
    const int tN = tile - tM * tilesN;
    const int row = (tM << 4) + lane15;
    const int col = (tN << 4) + lane15;

    const __bf16* brow = Bt + (size_t)col * K;
    v8f acc = {};

    for (int kk = 0; kk < (K >> 5); ++kk) {
        const int k0 = (kk << 5) + (kh << 3);    // A-fragment K base (runs of 8)
        v16bf a;
        if (A_BF16) {
            const __bf16* arow = (const __bf16*)Av + (size_t)row * K + k0;
            #pragma unroll
            for (int j = 0; j < 8; ++j) { a[j] = arow[j]; a[j + 8] = arow[j + 16]; }
        } else {
            const float* arow = (const float*)Av + (size_t)row * K + k0;
            #pragma unroll
            for (int j = 0; j < 8; ++j) { a[j] = (__bf16)arow[j]; a[j + 8] = (__bf16)arow[j + 16]; }
        }
        v16bf b;
        #pragma unroll
        for (int j = 0; j < 8; ++j) { b[j] = brow[k0 + j]; b[j + 8] = brow[k0 + 16 + j]; }
        acc = __builtin_amdgcn_wmma_f32_16x16x32_bf16(false, a, false, b,
                                                      (short)0, acc, false, false);
    }

    const float bc = HAS_BIAS ? bias[col] : 0.0f;
    #pragma unroll
    for (int r = 0; r < 8; ++r) {
        float v = acc[r] + bc;
        if (DO_TANH) v = fast_tanh(v);
        const size_t o = (size_t)((tM << 4) + r + (kh << 3)) * N + col;
        if (OUT_BF16) ((__bf16*)Dv)[o] = (__bf16)v;
        else          ((float*)Dv)[o]  = v;
    }
}

// ---------------------------------------------------------------------------
// Fused per-node kernel. Each 128-thread block processes 8 nodes; the filter
// weights (W_f1^T, W_f2^T bf16) are staged in LDS ONCE per block (8x cut in
// L2 weight traffic). Padded LDS row strides (72 / 136 elements) make the
// per-lane WMMA fragment reads bank-conflict-free while keeping 16B alignment.
// Per node:
//   H1      = tanh(rbf[32,64] @ W_f1 + b_f1)    (WMMA, A/B from LDS)
//   filters = H1 @ W_f2 + b_f2                  (WMMA, A/B from LDS)
//   filtered= gather(init_out) * filters
//   attn    = softmax(filtered @ nbr_filter)    (shfl + ds_add_f32)
//   conv    = attn^T @ filtered
// ---------------------------------------------------------------------------
#define SW1  72    // padded K-stride for W_f1^T rows (K=64)
#define SW2 136    // padded K-stride for W_f2^T rows (K=128)
#define SH  136    // padded col-stride for H1 / Nbr rows (F=128)
#define SR   72    // padded col-stride for rbf rows (G=64)

__global__ void __launch_bounds__(128)
fused_filter_attn(const float* __restrict__ rbf, const long long* __restrict__ nlist,
                  const float* __restrict__ b_f1, const float* __restrict__ b_f2,
                  const float* __restrict__ nbr_filter,
                  const __bf16* __restrict__ Wf1T, const __bf16* __restrict__ Wf2T,
                  const __bf16* __restrict__ init_out,
                  float* __restrict__ attn_out, float* __restrict__ conv_out)
{
    __shared__ __attribute__((aligned(32))) __bf16 sWf1[128 * SW1];  // 18.0 KB
    __shared__ __attribute__((aligned(32))) __bf16 sWf2[128 * SW2];  // 34.0 KB
    __shared__ __attribute__((aligned(32))) __bf16 sRbf[32 * SR];    //  4.5 KB
    __shared__ __attribute__((aligned(32))) __bf16 sH1 [32 * SH];    //  8.5 KB
    __shared__ __attribute__((aligned(32))) __bf16 sNbr[32 * SH];    //  8.5 KB
    __shared__ float sScore[32];
    __shared__ float sAttn[32];

    const int tid    = threadIdx.x;
    const int wave   = tid >> 5;
    const int lane   = tid & 31;
    const int lane15 = lane & 15;
    const int kh     = lane >> 4;

    // ---- stage both filter weights into LDS once per block ----
    for (int c = tid; c < 1024; c += 128) {                  // W_f1^T: 128x64
        v8bf v = *(const v8bf*)(Wf1T + c * 8);
        *(v8bf*)(sWf1 + (c >> 3) * SW1 + (c & 7) * 8) = v;
    }
    for (int c = tid; c < 2048; c += 128) {                  // W_f2^T: 128x128
        v8bf v = *(const v8bf*)(Wf2T + c * 8);
        *(v8bf*)(sWf2 + (c >> 4) * SW2 + (c & 15) * 8) = v;
    }

    float nfc[2], bc1[2], bc2[2];
    #pragma unroll
    for (int n = 0; n < 2; ++n) {
        const int col = wave * 32 + n * 16 + lane15;
        nfc[n] = nbr_filter[col];
        bc1[n] = b_f1[col];
        bc2[n] = b_f2[col];
    }

    for (int it = 0; it < 8; ++it) {
        const int node = (blockIdx.x << 3) + it;
        __syncthreads();          // weights staged / previous iteration done
        if (tid < 32) sScore[tid] = 0.0f;

        // prefetch next node's rbf tile (global_prefetch_b8)
        if (it < 7) __builtin_prefetch(rbf + (size_t)(node + 1) * 2048 + tid * 16, 0, 0);

        // ---- stage rbf tile [32,64] f32 -> bf16 LDS (padded rows) ----
        const float* rb = rbf + (size_t)node * 2048;
        #pragma unroll
        for (int i = 0; i < 16; ++i) {
            const int e = tid + i * 128;
            sRbf[(e >> 6) * SR + (e & 63)] = (__bf16)rb[e];
        }

        // ---- gather 32 neighbor feature rows (bf16) into LDS ----
        {
            const int r = tid >> 2, q = tid & 3;
            const long long nb = nlist[(size_t)node * 32 + r];
            const int gbase = (node >> 10) << 10;            // b * N
            const __bf16* src = init_out + ((size_t)(gbase + (int)nb)) * 128 + q * 32;
            __bf16* dst = sNbr + r * SH + q * 32;
            *(v8bf*)(dst + 0)  = *(const v8bf*)(src + 0);
            *(v8bf*)(dst + 8)  = *(const v8bf*)(src + 8);
            *(v8bf*)(dst + 16) = *(const v8bf*)(src + 16);
            *(v8bf*)(dst + 24) = *(const v8bf*)(src + 24);
        }
        __syncthreads();

        v8f acc[2][2] = {};

        // ---- GEMM1: [32,64] @ [64, 32-per-wave], all operands in LDS ----
        #pragma unroll
        for (int kk = 0; kk < 2; ++kk) {
            v16bf af[2];
            #pragma unroll
            for (int m = 0; m < 2; ++m) {
                const __bf16* ar = sRbf + (m * 16 + lane15) * SR + kk * 32 + kh * 8;
                #pragma unroll
                for (int j = 0; j < 8; ++j) { af[m][j] = ar[j]; af[m][j + 8] = ar[j + 16]; }
            }
            #pragma unroll
            for (int n = 0; n < 2; ++n) {
                const int col = wave * 32 + n * 16 + lane15;
                const __bf16* bp = sWf1 + col * SW1 + kk * 32 + kh * 8;
                v16bf bf;
                #pragma unroll
                for (int j = 0; j < 8; ++j) { bf[j] = bp[j]; bf[j + 8] = bp[j + 16]; }
                #pragma unroll
                for (int m = 0; m < 2; ++m)
                    acc[m][n] = __builtin_amdgcn_wmma_f32_16x16x32_bf16(
                        false, af[m], false, bf, (short)0, acc[m][n], false, false);
            }
        }

        // ---- tanh epilogue -> sH1 bf16 (branch-free fast tanh) ----
        #pragma unroll
        for (int n = 0; n < 2; ++n) {
            const int col = wave * 32 + n * 16 + lane15;
            #pragma unroll
            for (int m = 0; m < 2; ++m)
                #pragma unroll
                for (int r = 0; r < 8; ++r) {
                    const int R = m * 16 + r + kh * 8;
                    sH1[R * SH + col] = (__bf16)fast_tanh(acc[m][n][r] + bc1[n]);
                }
        }
        __syncthreads();

        // ---- GEMM2: [32,128] @ [128, 32-per-wave], all operands in LDS ----
        #pragma unroll
        for (int m = 0; m < 2; ++m)
            #pragma unroll
            for (int n = 0; n < 2; ++n) { v8f z = {}; acc[m][n] = z; }

        #pragma unroll
        for (int kk = 0; kk < 4; ++kk) {
            v16bf af[2];
            #pragma unroll
            for (int m = 0; m < 2; ++m) {
                const __bf16* ar = sH1 + (m * 16 + lane15) * SH + kk * 32 + kh * 8;
                #pragma unroll
                for (int j = 0; j < 8; ++j) { af[m][j] = ar[j]; af[m][j + 8] = ar[j + 16]; }
            }
            #pragma unroll
            for (int n = 0; n < 2; ++n) {
                const int col = wave * 32 + n * 16 + lane15;
                const __bf16* bp = sWf2 + col * SW2 + kk * 32 + kh * 8;
                v16bf bf;
                #pragma unroll
                for (int j = 0; j < 8; ++j) { bf[j] = bp[j]; bf[j + 8] = bp[j + 16]; }
                #pragma unroll
                for (int m = 0; m < 2; ++m)
                    acc[m][n] = __builtin_amdgcn_wmma_f32_16x16x32_bf16(
                        false, af[m], false, bf, (short)0, acc[m][n], false, false);
            }
        }

        // ---- filtered = (filters + b_f2) * nbr ; per-row score partials ----
        #pragma unroll
        for (int m = 0; m < 2; ++m) {
            #pragma unroll
            for (int r = 0; r < 8; ++r) {
                const int R = m * 16 + r + kh * 8;
                float s = 0.0f;
                #pragma unroll
                for (int n = 0; n < 2; ++n) {
                    const int col = wave * 32 + n * 16 + lane15;
                    const float f = (acc[m][n][r] + bc2[n]) * (float)sNbr[R * SH + col];
                    acc[m][n][r] = f;
                    s += f * nfc[n];
                }
                // reduce across the 16 lanes of each half (xor 1..8 keeps halves apart)
                s += __shfl_xor(s, 1);
                s += __shfl_xor(s, 2);
                s += __shfl_xor(s, 4);
                s += __shfl_xor(s, 8);
                if (lane15 == 0) atomicAdd(&sScore[R], s);   // ds_add_f32
            }
        }
        __syncthreads();

        // ---- softmax over K=32 neighbors (one wave) ----
        if (wave == 0) {
            const float sc = sScore[lane];
            float mx = sc;
            mx = fmaxf(mx, __shfl_xor(mx, 1));
            mx = fmaxf(mx, __shfl_xor(mx, 2));
            mx = fmaxf(mx, __shfl_xor(mx, 4));
            mx = fmaxf(mx, __shfl_xor(mx, 8));
            mx = fmaxf(mx, __shfl_xor(mx, 16));
            const float e = __expf(sc - mx);
            float sum = e;
            sum += __shfl_xor(sum, 1);
            sum += __shfl_xor(sum, 2);
            sum += __shfl_xor(sum, 4);
            sum += __shfl_xor(sum, 8);
            sum += __shfl_xor(sum, 16);
            const float a = e / sum;
            sAttn[lane] = a;
            attn_out[(size_t)node * 32 + lane] = a;
        }
        __syncthreads();

        // ---- conv_out[col] = sum_R attn[R] * filtered[R,col] ----
        #pragma unroll
        for (int n = 0; n < 2; ++n) {
            const int col = wave * 32 + n * 16 + lane15;
            float c = 0.0f;
            #pragma unroll
            for (int m = 0; m < 2; ++m)
                #pragma unroll
                for (int r = 0; r < 8; ++r)
                    c += sAttn[m * 16 + r + kh * 8] * acc[m][n][r];
            c += __shfl_xor(c, 16);           // merge row halves (same column)
            if (kh == 0) conv_out[(size_t)node * 128 + col] = c;
        }
    }
}

// ---------------------------------------------------------------------------
// Host-side orchestration
// ---------------------------------------------------------------------------
extern "C" void kernel_launch(void* const* d_in, const int* in_sizes, int n_in,
                              void* d_out, int out_size, void* d_ws, size_t ws_size,
                              hipStream_t stream) {
    const float*     features = (const float*)d_in[0];
    const float*     rbf      = (const float*)d_in[1];
    const long long* nlist    = (const long long*)d_in[2];   // int64 per reference
    const float*     W_init   = (const float*)d_in[3];
    const float*     W_f1     = (const float*)d_in[4];
    const float*     b_f1     = (const float*)d_in[5];
    const float*     W_f2     = (const float*)d_in[6];
    const float*     b_f2     = (const float*)d_in[7];
    const float*     nbrf     = (const float*)d_in[8];
    const float*     W_o1     = (const float*)d_in[9];
    const float*     b_o1     = (const float*)d_in[10];
    const float*     W_o2     = (const float*)d_in[11];
    const float*     b_o2     = (const float*)d_in[12];

    char* ws = (char*)d_ws;
    __bf16* WinitT   = (__bf16*)(ws + 0);        // 32 KB
    __bf16* Wf1T     = (__bf16*)(ws + 32768);    // 16 KB
    __bf16* Wf2T     = (__bf16*)(ws + 49152);    // 32 KB
    __bf16* Wo1T     = (__bf16*)(ws + 81920);    // 32 KB
    __bf16* Wo2T     = (__bf16*)(ws + 114688);   // 32 KB
    __bf16* init_out = (__bf16*)(ws + 147456);   // 8192*128 bf16 = 2 MB
    float*  conv     = (float*)(ws + 2244608);   // 8192*128 f32  = 4 MB
    __bf16* Hmid     = (__bf16*)(ws + 6438912);  // 8192*128 bf16 = 2 MB

    float* out  = (float*)d_out;                 // [8192,128]
    float* attn = out + (size_t)8192 * 128;      // [8192,32]

    // 1) weights -> transposed bf16
    convert_weights<<<288, 256, 0, stream>>>(W_init, W_f1, W_f2, W_o1, W_o2,
                                             WinitT, Wf1T, Wf2T, Wo1T, Wo2T);

    // 2) init_out = features @ W_init (bf16 out), M=8192 N=128 K=128
    gemm_wmma<false, true, false, false><<<512, 256, 0, stream>>>(
        features, WinitT, nullptr, init_out, 8192, 128, 128);

    // 3) fused filter-generator + gather + attention: 8 nodes per block
    fused_filter_attn<<<1024, 128, 0, stream>>>(
        rbf, nlist, b_f1, b_f2, nbrf, Wf1T, Wf2T, init_out, attn, conv);

    // 4) Hmid = tanh(conv @ W_o1 + b_o1) (bf16 out)
    gemm_wmma<false, true, true, true><<<512, 256, 0, stream>>>(
        conv, Wo1T, b_o1, Hmid, 8192, 128, 128);

    // 5) out = Hmid @ W_o2 + b_o2 (f32 out)
    gemm_wmma<true, false, false, true><<<512, 256, 0, stream>>>(
        Hmid, Wo2T, b_o2, (void*)out, 8192, 128, 128);
}
